// MultiHeadAttention_65910568125151
// MI455X (gfx1250) — compile-verified, matrix operand order checked by reading, not run on hardware
//
#include <hip/hip_runtime.h>

// ---------------- problem constants ----------------
constexpr int Bc = 4, Sc = 1024, Dc = 1024, Hc = 16, DKc = 64;

// ---------------- vector types ----------------
typedef __attribute__((ext_vector_type(16))) __bf16 v16bf;
typedef __attribute__((ext_vector_type(8)))  __bf16 v8bf;
typedef __attribute__((ext_vector_type(4)))  __bf16 v4bf;
typedef __attribute__((ext_vector_type(8)))  float  v8f;

// float -> bf16 via native hardware convert (v_cvt_pk_bf16_f32 on gfx1250)
static __device__ __forceinline__ __bf16 f2bf(float f) { return (__bf16)f; }

static __device__ __forceinline__ v8f wmma_bf(v16bf a, v16bf b, v8f c) {
    return __builtin_amdgcn_wmma_f32_16x16x32_bf16(false, a, false, b, (short)0, c,
                                                   false, false);
}

// A/B fragment (16x32 bf16): elements 0..7 at p[0..7], 8..15 at p[16..23]
// (per CDNA5 16-bit matrix VGPR layout; both halves are 16B-aligned b128 loads).
static __device__ __forceinline__ v16bf frag_bf(const __bf16* p) {
    v8bf lo = *(const v8bf*)p;
    v8bf hi = *(const v8bf*)(p + 16);
    v16bf r;
#pragma unroll
    for (int i = 0; i < 8; ++i) { r[i] = lo[i]; r[8 + i] = hi[i]; }
    return r;
}

// ---------------- elementwise f32 -> bf16 convert (vectorized) ----------------
__global__ void cvt_bf16(const float* __restrict__ src, __bf16* __restrict__ dst,
                         int n4) {
    int i = blockIdx.x * blockDim.x + threadIdx.x;
    if (i < n4) {
        float4 x = ((const float4*)src)[i];
        v4bf o;
        o[0] = f2bf(x.x); o[1] = f2bf(x.y); o[2] = f2bf(x.z); o[3] = f2bf(x.w);
        ((v4bf*)dst)[i] = o;
    }
}

// ---------------- GEMM:  C[M,N] = X[M,K] @ W[N,K]^T + bias ----------------
// All operands bf16; inner loop is pure b128 loads + WMMA.
// MODE 0: store bf16 row-major [M,N]
// MODE 1: store bf16 transposed per-head: [B,H,DK,S]  (for V)
// MODE 2: store f32 row-major [M,N]  (final output)
template <int MODE>
__global__ __launch_bounds__(128) void gemm_proj(const __bf16* __restrict__ X,
                                                 const __bf16* __restrict__ W,
                                                 const float* __restrict__ bias,
                                                 void* __restrict__ outv,
                                                 int N, int K) {
    const int lane = threadIdx.x & 31;
    const int wave = threadIdx.x >> 5;
    const int l16  = lane & 15;
    const int g    = lane >> 4;
    const int row0 = blockIdx.y * 64 + wave * 16;
    const int col0 = blockIdx.x * 64;

    v8f acc[4];
#pragma unroll
    for (int nc = 0; nc < 4; ++nc)
#pragma unroll
        for (int r = 0; r < 8; ++r) acc[nc][r] = 0.0f;

    for (int kk = 0; kk < K; kk += 32) {
        const v16bf a = frag_bf(X + (size_t)(row0 + l16) * K + kk + 8 * g);
#pragma unroll
        for (int nc = 0; nc < 4; ++nc) {
            const v16bf bfrag =
                frag_bf(W + (size_t)(col0 + nc * 16 + l16) * K + kk + 8 * g);
            acc[nc] = wmma_bf(a, bfrag, acc[nc]);
        }
    }

#pragma unroll
    for (int nc = 0; nc < 4; ++nc) {
#pragma unroll
        for (int r = 0; r < 8; ++r) {
            const int m = row0 + r + 8 * g;
            const int n = col0 + nc * 16 + l16;
            const float val = acc[nc][r] + bias[n];
            if constexpr (MODE == 0) {
                ((__bf16*)outv)[(size_t)m * N + n] = f2bf(val);
            } else if constexpr (MODE == 1) {
                const int bb = m >> 10, s = m & 1023;   // m = b*S + s, S=1024
                const int hh = n >> 6,  dk = n & 63;    // n = h*64 + dk
                ((__bf16*)outv)[(((size_t)(bb * Hc + hh)) * DKc + dk) * Sc + s] = f2bf(val);
            } else {
                ((float*)outv)[(size_t)m * N + n] = val;
            }
        }
    }
}

// ---------------- flash attention (per-head, online softmax) ----------------
// qb, kb: bf16 [B,S,D];  vtb: bf16 [B,H,DK,S];  attnb: bf16 [B,S,D]
__global__ __launch_bounds__(128) void attn_kernel(const __bf16* __restrict__ qb,
                                                   const __bf16* __restrict__ kb,
                                                   const __bf16* __restrict__ vtb,
                                                   const float* __restrict__ prob,
                                                   const int* __restrict__ mask,
                                                   const int* __restrict__ lam_ptr,
                                                   __bf16* __restrict__ attnb) {
    __shared__ float pbuf[4][16 * 33];   // per-wave P tile (padded stride)
    __shared__ float probS[4][32 * 16];  // per-wave async-staged prob^T tile
    const int lane = threadIdx.x & 31;
    const int wave = threadIdx.x >> 5;
    const int l16  = lane & 15;
    const int g    = lane >> 4;
    const int b = blockIdx.z, h = blockIdx.y;
    const int q0 = (blockIdx.x * 4 + wave) * 16;
    const float lamf = (float)lam_ptr[0];

    // Q tile A-fragments (16 x 64, split into two K=32 chunks)
    const __bf16* qrow = qb + (size_t)(b * Sc + q0 + l16) * Dc + h * DKc;
    const v16bf aq0 = frag_bf(qrow + 8 * g);
    const v16bf aq1 = frag_bf(qrow + 32 + 8 * g);

    v8f o[4];
    float mrow[8], lrow[8];
#pragma unroll
    for (int r = 0; r < 8; ++r) {
        mrow[r] = -3.0e38f; lrow[r] = 0.0f;
#pragma unroll
        for (int nc = 0; nc < 4; ++nc) o[nc][r] = 0.0f;
    }

    for (int kt = 0; kt < Sc; kt += 32) {
        // ---- async-stage prob[b, kt..kt+31, q0..q0+15] into LDS ----------
        // Each b128 moves 8 rows: 4 consecutive lanes cover one 64B row
        // (one cacheline), overlapping with the QK^T WMMAs below (ASYNCcnt).
#pragma unroll
        for (int i = 0; i < 4; ++i) {
            const int row = (lane >> 2) + i * 8;          // prob row (key) in tile
            const int c4  = (lane & 3) * 4;               // float chunk in row
            const float* gsrc =
                prob + ((size_t)(b * Sc + kt + row) * Sc + q0) + c4;
            unsigned ldst = (unsigned)(uintptr_t)&probS[wave][row * 16 + c4];
            asm volatile("global_load_async_to_lds_b128 %0, %1, off"
                         :: "v"(ldst), "v"(gsrc) : "memory");
        }

        if (kt + 32 < Sc)  // prefetch next key tile (global_prefetch)
            __builtin_prefetch(kb + (size_t)(b * Sc + kt + 32 + l16) * Dc + h * DKc, 0, 1);

        // ---- scores: two 16x16 tiles (keys kt..+15, kt+16..+31) ----
        v8f s[2];
#pragma unroll
        for (int t = 0; t < 2; ++t) {
            const int key = kt + t * 16 + l16;
            const __bf16* krow = kb + (size_t)(b * Sc + key) * Dc + h * DKc;
            v16bf b0 = frag_bf(krow + 8 * g);
            v16bf b1 = frag_bf(krow + 32 + 8 * g);
            v8f c;
#pragma unroll
            for (int r = 0; r < 8; ++r) c[r] = 0.0f;
            c = wmma_bf(aq0, b0, c);
            c = wmma_bf(aq1, b1, c);
            s[t] = c;
        }

        // wait for the async prob tile, then apply scale/bias/mask
        asm volatile("s_wait_asynccnt 0x0" ::: "memory");
#pragma unroll
        for (int t = 0; t < 2; ++t) {
#pragma unroll
            for (int r = 0; r < 8; ++r) {
                const int m = q0 + r + 8 * g;
                const int key = kt + t * 16 + l16;
                float sc = s[t][r] * 0.125f
                         - lamf * probS[wave][(t * 16 + l16) * 16 + (r + 8 * g)];
                if (mask[((size_t)b * Sc + m) * Sc + key] == 0) sc = -1e9f;
                s[t][r] = sc;
            }
        }

        // ---- online softmax (16-lane butterflies stay inside half-wave) ----
#pragma unroll
        for (int r = 0; r < 8; ++r) {
            float tm = fmaxf(s[0][r], s[1][r]);
#pragma unroll
            for (int d = 1; d < 16; d <<= 1) tm = fmaxf(tm, __shfl_xor(tm, d, 32));
            const float mn   = fmaxf(mrow[r], tm);
            const float corr = __expf(mrow[r] - mn);
            const float p0 = __expf(s[0][r] - mn);
            const float p1 = __expf(s[1][r] - mn);
            float ps = p0 + p1;
#pragma unroll
            for (int d = 1; d < 16; d <<= 1) ps += __shfl_xor(ps, d, 32);
            lrow[r] = lrow[r] * corr + ps;
            mrow[r] = mn;
#pragma unroll
            for (int nc = 0; nc < 4; ++nc) o[nc][r] *= corr;
            pbuf[wave][(r + 8 * g) * 33 + l16]      = p0;
            pbuf[wave][(r + 8 * g) * 33 + 16 + l16] = p1;
        }
        __syncthreads();

        // ---- transpose P (C-layout) to A-layout through LDS ----
        v16bf ap;
#pragma unroll
        for (int e = 0; e < 16; ++e) {
            const int kk = (e < 8) ? (8 * g + e) : (16 + 8 * g + (e - 8));
            ap[e] = f2bf(pbuf[wave][l16 * 33 + kk]);
        }
        __syncthreads();

        // ---- PV: 4 WMMAs over the transposed V layout ----
#pragma unroll
        for (int nc = 0; nc < 4; ++nc) {
            const __bf16* vrow =
                vtb + ((size_t)((b * Hc + h) * DKc + nc * 16 + l16)) * Sc + kt + 8 * g;
            v16bf bv = frag_bf(vrow);
            o[nc] = wmma_bf(ap, bv, o[nc]);
        }
    }

    // ---- normalize and write attn (bf16, [B,S,D] layout, heads merged) ----
#pragma unroll
    for (int nc = 0; nc < 4; ++nc) {
#pragma unroll
        for (int r = 0; r < 8; ++r) {
            const int m = q0 + r + 8 * g;
            const float v = o[nc][r] / lrow[r];
            attnb[(size_t)(b * Sc + m) * Dc + h * DKc + nc * 16 + l16] = f2bf(v);
        }
    }
}

// forward the mask (second tuple output) into the tail of d_out
__global__ void copy_i32(const int* __restrict__ src, int* __restrict__ dst, int n) {
    int i = blockIdx.x * blockDim.x + threadIdx.x;
    if (i < n) dst[i] = src[i];
}

extern "C" void kernel_launch(void* const* d_in, const int* in_sizes, int n_in,
                              void* d_out, int out_size, void* d_ws, size_t ws_size,
                              hipStream_t stream) {
    const float* Qx   = (const float*)d_in[0];
    const float* Kx   = (const float*)d_in[1];
    const float* Vx   = (const float*)d_in[2];
    const float* prob = (const float*)d_in[3];
    const int*   mask = (const int*)d_in[4];
    const int*   lam  = (const int*)d_in[5];
    const float* Wq = (const float*)d_in[6];
    const float* bq = (const float*)d_in[7];
    const float* Wk = (const float*)d_in[8];
    const float* bk = (const float*)d_in[9];
    const float* Wv = (const float*)d_in[10];
    const float* bv = (const float*)d_in[11];
    const float* Wo = (const float*)d_in[12];
    const float* bo = (const float*)d_in[13];

    char* ws = (char*)d_ws;
    const size_t tsz = (size_t)Bc * Sc * Dc * sizeof(__bf16);  // 8 MB
    const size_t wsz = (size_t)Dc * Dc * sizeof(__bf16);       // 2 MB
    __bf16* qb    = (__bf16*)(ws + 0 * tsz);
    __bf16* kb    = (__bf16*)(ws + 1 * tsz);
    __bf16* vtb   = (__bf16*)(ws + 2 * tsz);
    __bf16* attnb = (__bf16*)(ws + 3 * tsz);
    __bf16* xq    = (__bf16*)(ws + 4 * tsz);
    __bf16* xk    = (__bf16*)(ws + 5 * tsz);
    __bf16* xv    = (__bf16*)(ws + 6 * tsz);
    __bf16* wqb   = (__bf16*)(ws + 7 * tsz + 0 * wsz);
    __bf16* wkb   = (__bf16*)(ws + 7 * tsz + 1 * wsz);
    __bf16* wvb   = (__bf16*)(ws + 7 * tsz + 2 * wsz);
    __bf16* wob   = (__bf16*)(ws + 7 * tsz + 3 * wsz);

    float* outF    = (float*)d_out;
    int*   maskOut = (int*)(outF + (size_t)Bc * Sc * Dc);

    // ---- one-time f32 -> bf16 conversion passes ----
    const int nx4 = (Bc * Sc * Dc) / 4;   // activations, float4 per thread
    const int nw4 = (Dc * Dc) / 4;        // weights
    cvt_bf16<<<(nx4 + 255) / 256, 256, 0, stream>>>(Qx, xq, nx4);
    cvt_bf16<<<(nx4 + 255) / 256, 256, 0, stream>>>(Kx, xk, nx4);
    cvt_bf16<<<(nx4 + 255) / 256, 256, 0, stream>>>(Vx, xv, nx4);
    cvt_bf16<<<(nw4 + 255) / 256, 256, 0, stream>>>(Wq, wqb, nw4);
    cvt_bf16<<<(nw4 + 255) / 256, 256, 0, stream>>>(Wk, wkb, nw4);
    cvt_bf16<<<(nw4 + 255) / 256, 256, 0, stream>>>(Wv, wvb, nw4);
    cvt_bf16<<<(nw4 + 255) / 256, 256, 0, stream>>>(Wo, wob, nw4);

    // ---- projections ----
    const dim3 gg(Dc / 64, (Bc * Sc) / 64);  // (16, 64)
    gemm_proj<0><<<gg, 128, 0, stream>>>(xq, wqb, bq, qb, Dc, Dc);
    gemm_proj<0><<<gg, 128, 0, stream>>>(xk, wkb, bk, kb, Dc, Dc);
    gemm_proj<1><<<gg, 128, 0, stream>>>(xv, wvb, bv, vtb, Dc, Dc);

    // ---- attention ----
    attn_kernel<<<dim3(Sc / 64, Hc, Bc), 128, 0, stream>>>(qb, kb, vtb, prob, mask,
                                                           lam, attnb);

    // ---- output projection (f32 out) ----
    gemm_proj<2><<<gg, 128, 0, stream>>>(attnb, wob, bo, outF, Dc, Dc);

    // ---- forward mask output ----
    const int nmask = Bc * Sc * Sc;
    copy_i32<<<(nmask + 255) / 256, 256, 0, stream>>>(mask, maskOut, nmask);
}
